// Model_4449586118985
// MI455X (gfx1250) — matrix operand order repaired and checked by reading, not compiled, not run on hardware
//
#include <hip/hip_runtime.h>
#include <hip/hip_bf16.h>
#include <math.h>

// ---- CDNA5 WMMA types ----
typedef __attribute__((ext_vector_type(16))) __bf16 v16bf;
typedef __attribute__((ext_vector_type(8)))  float  v8f;
typedef __attribute__((ext_vector_type(4)))  int    i4;

struct I4Pair { i4 lo; i4 hi; };
union FragU { v16bf v; I4Pair p; };

#define Bn 32
#define Tn 1024
#define Fn 128
#define Hn 256
#define H4 1024   // 4*H

__device__ __forceinline__ unsigned short f2bf(float f) {
    unsigned int u = __float_as_uint(f);
    unsigned int r = u + 0x7FFFu + ((u >> 16) & 1u);   // RNE
    return (unsigned short)(r >> 16);
}

// ---------------------------------------------------------------------------
// Pack [W;U] -> bf16, column-major (col*K + k), gate-interleaved columns:
//   packed col pc: grp = pc>>6 (the owning wave), gate = (pc>>4)&3, lc = pc&15
//   original col  = gate*256 + grp*16 + lc
// Also packs bias into the same column order.
// ---------------------------------------------------------------------------
__global__ void pack_kernel(const float* __restrict__ W0, const float* __restrict__ U0, const float* __restrict__ b0,
                            const float* __restrict__ W1, const float* __restrict__ U1, const float* __restrict__ b1,
                            const float* __restrict__ W2, const float* __restrict__ U2, const float* __restrict__ b2,
                            unsigned short* __restrict__ Wp0,
                            unsigned short* __restrict__ Wp1,
                            unsigned short* __restrict__ Wp2,
                            float* __restrict__ bias_pk)
{
    int tid = blockIdx.x * blockDim.x + threadIdx.x;

    if (tid < 3 * H4) {                       // bias, packed order
        int l = tid >> 10, pc = tid & 1023;
        int oc = ((pc >> 4) & 3) * Hn + ((pc >> 6) << 4) + (pc & 15);
        const float* bb = (l == 0) ? b0 : ((l == 1) ? b1 : b2);
        bias_pk[tid] = bb[oc];
    }

    const int NW0 = H4 * 384;                 // layer 0: K = 128+256
    const int NW12 = H4 * 512;                // layers 1,2: K = 256+256

    if (tid < NW0) {
        int pc = tid / 384, k = tid % 384;
        int oc = ((pc >> 4) & 3) * Hn + ((pc >> 6) << 4) + (pc & 15);
        float v = (k < Fn) ? W0[k * H4 + oc] : U0[(k - Fn) * H4 + oc];
        Wp0[tid] = f2bf(v);
    }
    if (tid < NW12) {
        int pc = tid >> 9, k = tid & 511;
        int oc = ((pc >> 4) & 3) * Hn + ((pc >> 6) << 4) + (pc & 15);
        float v1 = (k < Hn) ? W1[k * H4 + oc] : U1[(k - Hn) * H4 + oc];
        float v2 = (k < Hn) ? W2[k * H4 + oc] : U2[(k - Hn) * H4 + oc];
        Wp1[tid] = f2bf(v1);
        Wp2[tid] = f2bf(v2);
    }
}

// ---------------------------------------------------------------------------
// mask[(b*2+br)*T + t] = any(x[b,br,t,:] != 0) ? 1.0 : 0.0
// ---------------------------------------------------------------------------
__global__ void mask_kernel(const float* __restrict__ x, float* __restrict__ maskf)
{
    int idx = blockIdx.x * blockDim.x + threadIdx.x;
    if (idx >= Bn * 2 * Tn) return;
    const float* p = x + (size_t)idx * Fn;
    float any = 0.f;
    for (int f = 0; f < Fn; ++f) any = (p[f] != 0.f) ? 1.f : any;
    maskf[idx] = any;
}

// ---------------------------------------------------------------------------
// Recurrent kernel: one workgroup per branch (16 waves, 512 threads).
// Wave w owns packed cols [w*64, w*64+64) = 16 h-cols x 4 gates.
// c/h state for those 16 h-cols x 32 batch rows lives in VGPRs in the
// WMMA C/D layout; h is re-published to LDS (bf16) each step.
// (Single B buffer: double-buffering spilled to scratch — see round 2.)
// ---------------------------------------------------------------------------
__global__ __launch_bounds__(512, 1) void lstm_kernel(
    const float* __restrict__ x,
    const unsigned short* __restrict__ Wp0,
    const unsigned short* __restrict__ Wp1,
    const unsigned short* __restrict__ Wp2,
    const float* __restrict__ bias_pk,
    const float* __restrict__ maskf,
    float* __restrict__ hfinal)
{
    __shared__ __align__(16) unsigned short sh_x[Bn * Fn];       // 8 KB
    __shared__ __align__(16) unsigned short sh_h[3][Bn * Hn];    // 48 KB
    __shared__ float sh_m[Bn];

    const int br   = blockIdx.x;
    const int tid  = threadIdx.x;
    const int wave = tid >> 5;
    const int lane = tid & 31;
    const int lh   = lane & 15;          // N column within tile / row within half
    const int hi16 = (lane >> 4) & 1;
    const int rowbase = hi16 * 8;        // C/D: lanes 16-31 hold M = v+8

    for (int i = tid; i < 3 * Bn * Hn; i += 512) (&sh_h[0][0])[i] = 0;

    // persistent per-wave state
    float creg[3][2][8];
    float hreg[3][2][8];
    #pragma unroll
    for (int l = 0; l < 3; ++l)
        #pragma unroll
        for (int mt = 0; mt < 2; ++mt)
            #pragma unroll
            for (int v = 0; v < 8; ++v) { creg[l][mt][v] = 0.f; hreg[l][mt][v] = 0.f; }

    float biasv[3][4];
    #pragma unroll
    for (int l = 0; l < 3; ++l)
        #pragma unroll
        for (int nt = 0; nt < 4; ++nt)
            biasv[l][nt] = bias_pk[l * H4 + wave * 64 + nt * 16 + lh];

    const unsigned short* const Wps[3] = { Wp0, Wp1, Wp2 };
    const int Kl[3] = { 384, 512, 512 };
    const int Dl[3] = { 128, 256, 256 };

    __syncthreads();

    for (int t = 0; t < Tn; ++t) {
        // ---- stage x_t (f32 -> bf16) and mask into LDS ----
        {
            int row = tid >> 4;
            int f0  = (tid & 15) * 8;
            const float* xp = x + (((size_t)row * 2 + br) * Tn + t) * Fn + f0;
            unsigned int p0 = f2bf(xp[0]) | ((unsigned int)f2bf(xp[1]) << 16);
            unsigned int p1 = f2bf(xp[2]) | ((unsigned int)f2bf(xp[3]) << 16);
            unsigned int p2 = f2bf(xp[4]) | ((unsigned int)f2bf(xp[5]) << 16);
            unsigned int p3 = f2bf(xp[6]) | ((unsigned int)f2bf(xp[7]) << 16);
            i4 pk = (i4){ (int)p0, (int)p1, (int)p2, (int)p3 };
            *(i4*)&sh_x[row * Fn + f0] = pk;
            __builtin_prefetch((const void*)(xp + Fn), 0, 1);   // next timestep's x row
        }
        if (tid < Bn) sh_m[tid] = maskf[((size_t)tid * 2 + br) * Tn + t];
        __syncthreads();

        #pragma unroll
        for (int l = 0; l < 3; ++l) {
            const int K = Kl[l], D = Dl[l];
            const unsigned short* wbase = Wps[l];

            // accumulators seeded with bias (bias is per-column = per N-tile)
            v8f acc[2][4];
            #pragma unroll
            for (int mt = 0; mt < 2; ++mt)
                #pragma unroll
                for (int nt = 0; nt < 4; ++nt) {
                    v8f a;
                    #pragma unroll
                    for (int e = 0; e < 8; ++e) a[e] = biasv[l][nt];
                    acc[mt][nt] = a;
                }

            const int kblocks = K >> 5;
            #pragma unroll 2
            for (int kb = 0; kb < kblocks; ++kb) {
                const int k0 = kb * 32;
                // A source: [x_t | h_{l-1}] for k<D comes from layer input,
                // k>=D comes from this layer's own recurrent h.
                const unsigned short* asrc;
                int astride, klocal;
                if (k0 < D) {
                    asrc = (l == 0) ? sh_x : sh_h[l - 1];
                    astride = (l == 0) ? Fn : Hn;
                    klocal = k0;
                } else {
                    asrc = sh_h[l];
                    astride = Hn;
                    klocal = k0 - D;
                }
                // A frags: lane row = lh (+16 for 2nd M-tile);
                // K chunks {0..7,16..23} (lanes 0-15) / {8..15,24..31} (lanes 16-31)
                const int koffA = hi16 * 8;
                FragU a0, a1;
                a0.p.lo = *(const i4*)&asrc[lh        * astride + klocal + koffA];
                a0.p.hi = *(const i4*)&asrc[lh        * astride + klocal + koffA + 16];
                a1.p.lo = *(const i4*)&asrc[(lh + 16) * astride + klocal + koffA];
                a1.p.hi = *(const i4*)&asrc[(lh + 16) * astride + klocal + koffA + 16];

                #pragma unroll
                for (int nt = 0; nt < 4; ++nt) {
                    // B frag: lane col = lh; K {0..7,16..23} (lanes 0-15) / {8..15,24..31} (lanes 16-31)
                    const int col = wave * 64 + nt * 16 + lh;
                    const unsigned short* wp = wbase + (size_t)col * K + k0 + hi16 * 8;
                    FragU bf;
                    bf.p.lo = *(const i4*)&wp[0];
                    bf.p.hi = *(const i4*)&wp[16];
                    __builtin_prefetch((const void*)(wp + 32), 0, 1);  // next k-block

                    acc[0][nt] = __builtin_amdgcn_wmma_f32_16x16x32_bf16(
                        false, a0.v, false, bf.v, (short)0, acc[0][nt], false, false);
                    acc[1][nt] = __builtin_amdgcn_wmma_f32_16x16x32_bf16(
                        false, a1.v, false, bf.v, (short)0, acc[1][nt], false, false);
                }
            }
            __syncthreads();   // all waves done reading h for this layer's GEMM

            // ---- gates + state update (all in registers) ----
            #pragma unroll
            for (int mt = 0; mt < 2; ++mt) {
                #pragma unroll
                for (int v = 0; v < 8; ++v) {
                    float zi = acc[mt][0][v];
                    float zf = acc[mt][1][v];
                    float zg = acc[mt][2][v];
                    float zo = acc[mt][3][v];
                    float co = creg[l][mt][v];
                    float ho = hreg[l][mt][v];
                    float si = 1.f / (1.f + __expf(-zi));
                    float sf = 1.f / (1.f + __expf(-zf));
                    float so = 1.f / (1.f + __expf(-zo));
                    float cn = sf * co + si * tanhf(zg);
                    float hn = so * tanhf(cn);
                    int row = mt * 16 + rowbase + v;
                    float m = sh_m[row];                 // 1.0 run / 0.0 carry
                    float cu = co + m * (cn - co);
                    float hu = ho + m * (hn - ho);
                    creg[l][mt][v] = cu;
                    hreg[l][mt][v] = hu;
                    sh_h[l][row * Hn + wave * 16 + lh] = f2bf(hu);
                }
            }
            __syncthreads();   // new h visible to all waves
        }
    }

    // final h of layer 2 (masking-carried) -> workspace
    #pragma unroll
    for (int mt = 0; mt < 2; ++mt)
        #pragma unroll
        for (int v = 0; v < 8; ++v) {
            int row = mt * 16 + rowbase + v;
            hfinal[((size_t)br * Bn + row) * Hn + wave * 16 + lh] = hreg[2][mt][v];
        }
}

// ---------------------------------------------------------------------------
// Head: out[b] = sigmoid( clip(||h1-h2||, 1e-7, 1e7) * w + b )
// ---------------------------------------------------------------------------
__global__ void head_kernel(const float* __restrict__ hfinal,
                            const float* __restrict__ dw,
                            const float* __restrict__ db,
                            float* __restrict__ out)
{
    int b = threadIdx.x;
    if (b >= Bn) return;
    float s = 0.f;
    for (int j = 0; j < Hn; ++j) {
        float d = hfinal[b * Hn + j] - hfinal[(Bn + b) * Hn + j];
        s += d * d;
    }
    float dist = sqrtf(s);
    dist = fminf(fmaxf(dist, 1e-7f), 1e7f);
    float z = dist * dw[0] + db[0];
    out[b] = 1.f / (1.f + __expf(-z));
}

// ---------------------------------------------------------------------------
extern "C" void kernel_launch(void* const* d_in, const int* in_sizes, int n_in,
                              void* d_out, int out_size, void* d_ws, size_t ws_size,
                              hipStream_t stream)
{
    const float* x  = (const float*)d_in[0];
    const float* W0 = (const float*)d_in[1];
    const float* U0 = (const float*)d_in[2];
    const float* b0 = (const float*)d_in[3];
    const float* W1 = (const float*)d_in[4];
    const float* U1 = (const float*)d_in[5];
    const float* b1 = (const float*)d_in[6];
    const float* W2 = (const float*)d_in[7];
    const float* U2 = (const float*)d_in[8];
    const float* b2 = (const float*)d_in[9];
    const float* dw = (const float*)d_in[10];
    const float* db = (const float*)d_in[11];
    float* out = (float*)d_out;

    char* ws = (char*)d_ws;
    size_t off = 0;
    auto take = [&](size_t n) { char* p = ws + off; off = (off + n + 255) & ~(size_t)255; return p; };

    unsigned short* Wp0     = (unsigned short*)take((size_t)H4 * 384 * 2);
    unsigned short* Wp1     = (unsigned short*)take((size_t)H4 * 512 * 2);
    unsigned short* Wp2     = (unsigned short*)take((size_t)H4 * 512 * 2);
    float*          bias_pk = (float*)take(3 * H4 * sizeof(float));
    float*          maskf   = (float*)take((size_t)Bn * 2 * Tn * sizeof(float));
    float*          hfinal  = (float*)take((size_t)2 * Bn * Hn * sizeof(float));
    (void)ws_size; (void)in_sizes; (void)n_in; (void)out_size;

    {   // weight/bias pack: cover max(H4*512) elements
        int total = H4 * 512;
        pack_kernel<<<(total + 255) / 256, 256, 0, stream>>>(
            W0, U0, b0, W1, U1, b1, W2, U2, b2, Wp0, Wp1, Wp2, bias_pk);
    }
    {   // mask
        int total = Bn * 2 * Tn;
        mask_kernel<<<(total + 255) / 256, 256, 0, stream>>>(x, maskf);
    }
    // recurrent core: one workgroup per branch
    lstm_kernel<<<dim3(2), dim3(512), 0, stream>>>(x, Wp0, Wp1, Wp2, bias_pk, maskf, hfinal);
    // head
    head_kernel<<<dim3(1), dim3(32), 0, stream>>>(hfinal, dw, db, out);
}
